// MyCheckerboardRQS_per_channel_72258529788185
// MI455X (gfx1250) — compile-verified, matrix operand order checked by reading, not run on hardware
//
#include <hip/hip_runtime.h>
#include <hip/hip_bf16.h>

typedef __attribute__((ext_vector_type(16))) _Float16 v16h;
typedef __attribute__((ext_vector_type(8)))  _Float16 v8h;
typedef __attribute__((ext_vector_type(8)))  float    v8f;

#define Bn   16
#define Cc   12
#define Hh   128
#define Ww   128
#define CIN1 16      // 12 masked-x + 4 cond
#define HID  64
#define OUTC 276
#define MPAD 288     // OUTC padded to 18 wmma tiles
#define MPADP 289    // pbuf row stride (floats), odd vs 64 banks
#define K1P  160     // 9 taps*16ch padded to 10 taps
#define K3   576     // 9 taps*64ch
#define STEPn 23
#define C12T 128     // conv12 block: 4 waves (1 M-tile each, 4 N-subtiles)
#define C3T  576     // conv3 block: 18 waves

__device__ inline v16h frag_from_g(const _Float16* p) {
    v8h lo = *(const v8h*)(p);
    v8h hi = *(const v8h*)(p + 16);
    v16h r;
#pragma unroll
    for (int i = 0; i < 8; ++i) { r[i] = lo[i]; r[i + 8] = hi[i]; }
    return r;
}

__device__ inline v16h frag_from_2(v8h lo, v8h hi) {
    v16h r;
#pragma unroll
    for (int i = 0; i < 8; ++i) { r[i] = lo[i]; r[i + 8] = hi[i]; }
    return r;
}

// ---------------- pack weights to f16 GEMM layout; init logdet ----------------
__global__ void pack_weights_kernel(const float* __restrict__ W1,
                                    const float* __restrict__ W2,
                                    const float* __restrict__ W3,
                                    const float* __restrict__ logdet,
                                    _Float16* __restrict__ W1p,
                                    _Float16* __restrict__ W2p,
                                    _Float16* __restrict__ W3p,
                                    float* __restrict__ outld) {
    int i = blockIdx.x * 256 + threadIdx.x;
    if (i < 16) outld[i] = logdet[i];
    if (i < HID * K1P) {                               // [64][160], k = tap*16+cin
        int m = i / K1P, k = i % K1P, tap = k >> 4, cin = k & 15;
        float v = (tap < 9) ? W1[(m * CIN1 + cin) * 9 + tap] : 0.0f;
        W1p[i] = (_Float16)v;
    } else if (i < HID * K1P + HID * HID) {            // [64][64]
        int j = i - HID * K1P;
        W2p[j] = (_Float16)W2[j];
    } else if (i < HID * K1P + HID * HID + MPAD * K3) { // [288][576], k = tap*64+cin
        int j = i - HID * K1P - HID * HID;
        int m = j / K3, k = j % K3, tap = k >> 6, cin = k & 63;
        float v = (m < OUTC && tap < 9) ? W3[((size_t)m * HID + cin) * 9 + tap] : 0.0f;
        W3p[j] = (_Float16)v;
    }
}

// ---------------- pack z = [mask*x , cond] in NHWC f16 ----------------
__global__ void pack_z_kernel(const float* __restrict__ x,
                              const float* __restrict__ cond,
                              _Float16* __restrict__ zg) {
    size_t i = (size_t)blockIdx.x * 256 + threadIdx.x;   // B*H*W*16 elements
    int c = (int)(i & 15);
    size_t pix = i >> 4;
    int w = (int)(pix & 127), h = (int)((pix >> 7) & 127), b = (int)(pix >> 14);
    float v;
    if (c < Cc) {   // frozen = x kept at (h+w) odd
        v = ((h + w) & 1) ? x[(((size_t)b * Cc + c) * Hh + h) * Ww + w] : 0.0f;
    } else {
        v = cond[(((size_t)b * 4 + (c - Cc)) * Hh + h) * Ww + w];
    }
    zg[i] = (_Float16)v;
}

// ---------------- fused conv1(3x3)+relu -> conv2(1x1)+relu, h1 stays in LDS ----
__global__ __launch_bounds__(C12T) void conv12_kernel(
    const _Float16* __restrict__ zg, const _Float16* __restrict__ W1p,
    const float* __restrict__ b1, const _Float16* __restrict__ W2p,
    const float* __restrict__ b2, _Float16* __restrict__ h2g) {
    __shared__ __align__(16) _Float16 zt[3 * 66 * CIN1];   // [row][col][cin sw]
    __shared__ __align__(16) _Float16 h1t[64 * HID];       // [px][ch sw]
    __shared__ __align__(16) _Float16 h2t[64 * HID];       // [px][ch]

    int blk = blockIdx.x;                    // B*H*2 blocks of 64 pixels
    int wblk = blk & 1, hh = (blk >> 1) & 127, b = blk >> 8;
    int w0 = wblk * 64;
    int tid = threadIdx.x;

    for (int g = tid; g < 3 * 66 * 2; g += C12T) {
        int row = g / 132, rem = g % 132, col = rem >> 1, g0 = rem & 1;
        int gh = hh + row - 1, gw = w0 + col - 1;
        v8h v = {};
        if (gh >= 0 && gh < Hh && gw >= 0 && gw < Ww)
            v = *(const v8h*)(zg + ((((size_t)b * Hh + gh) * Ww + gw) * CIN1 + g0 * 8));
        int gs = g0 ^ ((col >> 3) & 1);
        *(v8h*)(&zt[(row * 66 + col) * CIN1 + (gs << 3)]) = v;
    }
    __syncthreads();

    int lane = tid & 31;
    int mi = tid >> 5;                       // 0..3 M-tile
    int lm = lane & 15;
    int koff = (lane >= 16) ? 8 : 0;
    int koffb = koff >> 3;                   // 0 or 1

    // per-lane swizzled column bases (hoisted; swizzle group = koffb ^ colbit)
    int za[3];
#pragma unroll
    for (int kx = 0; kx < 3; ++kx)
        za[kx] = (lm + kx) * CIN1 + ((koffb ^ (((lm + kx) >> 3) & 1)) << 3);

    // ---- conv1: M=64, K=144 (pad 160), N=64 px ----
    const _Float16* a1row = W1p + (mi * 16 + lm) * K1P;
    v8f acc[4];
#pragma unroll
    for (int s = 0; s < 4; ++s) acc[s] = (v8f){};
#pragma unroll
    for (int c = 0; c < 5; ++c) {
        int kk = c * 32;
        v16h a = frag_from_g(a1row + kk + koff);
#pragma unroll
        for (int s = 0; s < 4; ++s) {
            v8h bvs[2];
#pragma unroll
            for (int gsel = 0; gsel < 2; ++gsel) {
                int tap = 2 * c + gsel;                 // compile-time
                int ky = tap / 3; if (ky > 2) ky = 2;   // pad tap 9: A is zero anyway
                int kx = tap - (tap / 3) * 3;
                bvs[gsel] = *(const v8h*)(&zt[(ky * 66 + s * 16) * CIN1 + za[kx]]);
            }
            v16h bf = frag_from_2(bvs[0], bvs[1]);
            acc[s] = __builtin_amdgcn_wmma_f32_16x16x32_f16(false, a, false, bf,
                                                            (short)0, acc[s], false, false);
        }
    }
    {
        int mb = mi * 16 + ((lane >= 16) ? 8 : 0);
#pragma unroll
        for (int s = 0; s < 4; ++s) {
            int n = s * 16 + (lane & 15);
#pragma unroll
            for (int r = 0; r < 8; ++r) {
                int mm = mb + r;
                float v = acc[s][r] + b1[mm];
                int sw = ((mm >> 3) ^ (n & 7)) << 3 | (mm & 7);
                h1t[n * HID + sw] = (_Float16)(v > 0.0f ? v : 0.0f);
            }
        }
    }
    __syncthreads();

    // per-lane bases for conv2 B reads: group = (2j)^koffb ^ (lm&7)
    int h1a[4];
#pragma unroll
    for (int j = 0; j < 4; ++j)
        h1a[j] = lm * HID + ((((j * 2) ^ koffb ^ (lm & 7)) & 7) << 3);

    // ---- conv2: M=64, K=64, N=64 px (B from swizzled h1t) ----
    const _Float16* a2row = W2p + (mi * 16 + lm) * HID;
    v8f acc2[4];
#pragma unroll
    for (int s = 0; s < 4; ++s) acc2[s] = (v8f){};
#pragma unroll
    for (int c = 0; c < 2; ++c) {
        int kk = c * 32;
        v16h a = frag_from_g(a2row + kk + koff);
#pragma unroll
        for (int s = 0; s < 4; ++s) {
            v8h b0 = *(const v8h*)(&h1t[s * 16 * HID + h1a[(2 * c + 0) & 3]]);
            v8h b1v = *(const v8h*)(&h1t[s * 16 * HID + h1a[(2 * c + 1) & 3]]);
            v16h bf = frag_from_2(b0, b1v);
            acc2[s] = __builtin_amdgcn_wmma_f32_16x16x32_f16(false, a, false, bf,
                                                             (short)0, acc2[s], false, false);
        }
    }
    {
        int mb = mi * 16 + ((lane >= 16) ? 8 : 0);
#pragma unroll
        for (int s = 0; s < 4; ++s) {
            int n = s * 16 + (lane & 15);
#pragma unroll
            for (int r = 0; r < 8; ++r) {
                int mm = mb + r;
                float v = acc2[s][r] + b2[mm];
                h2t[n * HID + mm] = (_Float16)(v > 0.0f ? v : 0.0f);
            }
        }
    }
    __syncthreads();

    for (int g = tid; g < 64 * 8; g += C12T) {
        int n = g >> 3, hg = (g & 7) * 8;
        *(v8h*)(h2g + (((size_t)b * Hh + hh) * Ww + (w0 + n)) * HID + hg) =
            *(v8h*)(&h2t[n * HID + hg]);
    }
}

// ---- conv3(3x3, 64->276) fused with RQS spline + logdet; 64 px / block -------
__global__ __launch_bounds__(C3T) void conv3_rqs_kernel(
    const _Float16* __restrict__ h2g, const _Float16* __restrict__ W3p,
    const float* __restrict__ b3, const float* __restrict__ xg,
    float* __restrict__ outx, float* __restrict__ outld) {
    __shared__ __align__(16) _Float16 t2[3 * 66 * HID];   // halo tile [row][col][ch sw]
    __shared__ __align__(16) float pbuf[64 * MPADP];      // params (padded rows)
    __shared__ float ladbuf[C3T];

    int blk = blockIdx.x;                    // B*H*2 blocks of 64 pixels
    int wblk = blk & 1, hh = (blk >> 1) & 127, b = blk >> 8;
    int w0 = wblk * 64;
    int tid = threadIdx.x;

    for (int g = tid; g < 3 * 66 * 8; g += C3T) {
        int row = g / 528, rem = g % 528, col = rem >> 3, g0 = rem & 7;
        int gh = hh + row - 1, gw = w0 + col - 1;
        v8h v = {};
        if (gh >= 0 && gh < Hh && gw >= 0 && gw < Ww)
            v = *(const v8h*)(h2g + ((((size_t)b * Hh + gh) * Ww + gw) * HID + g0 * 8));
        int gs = g0 ^ (col & 7);
        *(v8h*)(&t2[(row * 66 + col) * HID + (gs << 3)]) = v;
    }
    __syncthreads();

    int lane = tid & 31;
    int wv = tid >> 5;                        // 0..17, one M-tile each
    int lm = lane & 15;
    int koff = (lane >= 16) ? 8 : 0;
    int koffb = koff >> 3;
    const _Float16* arow = W3p + (size_t)(wv * 16 + lm) * K3;

    // per-lane swizzled B-read bases: vaddr[kx][j], j = (2c+gsel)&3
    // group read = ((2j)^koffb) ^ ((lm+kx)&7)   (== g0 ^ col&7 of the store swizzle)
    int vaddr[3][4];
#pragma unroll
    for (int kx = 0; kx < 3; ++kx) {
#pragma unroll
        for (int j = 0; j < 4; ++j) {
            int gs = ((j * 2) ^ koffb ^ ((lm + kx) & 7)) & 7;
            vaddr[kx][j] = (lm + kx) * HID + (gs << 3);
        }
    }

    v8f acc[4];
#pragma unroll
    for (int s = 0; s < 4; ++s) acc[s] = (v8f){};

#pragma unroll
    for (int ky = 0; ky < 3; ++ky) {
#pragma unroll
        for (int cc = 0; cc < 6; ++cc) {
            int c = ky * 6 + cc;              // K chunk index (k = tap*64+cin)
            int kk = c * 32;
            int kx = cc >> 1;                 // tap = ky*3 + (cc>>1)
            v16h a = frag_from_g(arow + kk + koff);
#pragma unroll
            for (int s = 0; s < 4; ++s) {
                const _Float16* bp = &t2[(ky * 66 + s * 16) * HID];
                v8h b0 = *(const v8h*)(bp + vaddr[kx][(2 * c + 0) & 3]);
                v8h b1v = *(const v8h*)(bp + vaddr[kx][(2 * c + 1) & 3]);
                v16h bf = frag_from_2(b0, b1v);
                acc[s] = __builtin_amdgcn_wmma_f32_16x16x32_f16(false, a, false, bf,
                                                                (short)0, acc[s], false, false);
            }
        }
    }
    {
        int n0 = lane & 15;
        int mb = wv * 16 + ((lane >= 16) ? 8 : 0);
#pragma unroll
        for (int s = 0; s < 4; ++s) {
#pragma unroll
            for (int r = 0; r < 8; ++r) {
                int mm = mb + r;
                pbuf[(s * 16 + n0) * MPADP + mm] = acc[s][r] + ((mm < OUTC) ? b3[mm] : 0.0f);
            }
        }
    }
    __syncthreads();

    // ---- RQS: 64 px * 12 channels = 768 items ----
    float ladsum = 0.0f;
    for (int it = tid; it < 64 * Cc; it += C3T) {
        int p = it / Cc, c = it - p * Cc;
        int wpx = w0 + p;
        const float* pr = &pbuf[p * MPADP + c * STEPn];
        float xv = xg[(((size_t)b * Cc + c) * Hh + hh) * Ww + wpx];
        bool active = (((hh + wpx) & 1) == 0);
        float xa = active ? xv : 0.0f;

        float cwA[9], chA[9], dv[9];
        {
            float mx = pr[0];
#pragma unroll
            for (int i = 1; i < 8; ++i) mx = fmaxf(mx, pr[i]);
            float e[8], ssum = 0.0f;
#pragma unroll
            for (int i = 0; i < 8; ++i) { e[i] = expf(pr[i] - mx); ssum += e[i]; }
            float inv = 1.0f / ssum, cs = 0.0f;
            cwA[0] = -3.0f;
#pragma unroll
            for (int i = 0; i < 8; ++i) {
                cs += 0.001f + (1.0f - 0.008f) * e[i] * inv;
                cwA[i + 1] = 6.0f * cs - 3.0f;
            }
            cwA[8] = 3.0f;
        }
        {
            float mx = pr[8];
#pragma unroll
            for (int i = 1; i < 8; ++i) mx = fmaxf(mx, pr[8 + i]);
            float e[8], ssum = 0.0f;
#pragma unroll
            for (int i = 0; i < 8; ++i) { e[i] = expf(pr[8 + i] - mx); ssum += e[i]; }
            float inv = 1.0f / ssum, cs = 0.0f;
            chA[0] = -3.0f;
#pragma unroll
            for (int i = 0; i < 8; ++i) {
                cs += 0.001f + (1.0f - 0.008f) * e[i] * inv;
                chA[i + 1] = 6.0f * cs - 3.0f;
            }
            chA[8] = 3.0f;
        }
        dv[0] = 1.0f; dv[8] = 1.0f;
#pragma unroll
        for (int i = 0; i < 7; ++i) {
            float u = pr[16 + i];
            float sp = (u > 20.0f) ? u : log1pf(expf(u));
            dv[i + 1] = 0.001f + sp;
        }

        float xc = fminf(3.0f, fmaxf(-3.0f, xa));
        bool inside = (xa >= -3.0f) && (xa <= 3.0f);
        int idx = 0;
#pragma unroll
        for (int i = 1; i < 9; ++i) if (xc >= cwA[i]) idx = i;
        if (idx > 7) idx = 7;

        float icw = cwA[0], iw = 1.0f, ich = chA[0], ih = 1.0f, d0 = dv[0], d1 = dv[1];
#pragma unroll
        for (int i = 0; i < 8; ++i) {
            if (idx == i) {
                icw = cwA[i]; iw = cwA[i + 1] - cwA[i];
                ich = chA[i]; ih = chA[i + 1] - chA[i];
                d0 = dv[i]; d1 = dv[i + 1];
            }
        }
        float theta = (xc - icw) / iw;
        float omt = 1.0f - theta;
        float t1 = theta * omt;
        float delta = ih / iw;
        float num = ih * (delta * theta * theta + d0 * t1);
        float den = delta + (d0 + d1 - 2.0f * delta) * t1;
        float y = ich + num / den;
        float dnum = delta * delta * (d1 * theta * theta + 2.0f * delta * t1 + d0 * omt * omt);
        float ladv = logf(dnum) - 2.0f * logf(den);

        float yout = inside ? y : xa;
        float res = (active ? 0.0f : xv) + yout;       // x_frozen + y
        outx[(((size_t)b * Cc + c) * Hh + hh) * Ww + wpx] = res;
        ladsum += inside ? ladv : 0.0f;
    }
    ladbuf[tid] = ladsum;
    __syncthreads();
    if (tid < 64) {
        float s = 0.0f;
        for (int i = tid; i < C3T; i += 64) s += ladbuf[i];
        ladbuf[tid] = s;
    }
    __syncthreads();
    if (tid == 0) {
        float s = 0.0f;
        for (int i = 0; i < 64; ++i) s += ladbuf[i];
        atomicAdd(&outld[b], s);
    }
}

extern "C" void kernel_launch(void* const* d_in, const int* in_sizes, int n_in,
                              void* d_out, int out_size, void* d_ws, size_t ws_size,
                              hipStream_t stream) {
    (void)in_sizes; (void)n_in; (void)out_size; (void)ws_size;
    const float* x      = (const float*)d_in[0];
    const float* logdet = (const float*)d_in[1];
    const float* cond   = (const float*)d_in[2];
    const float* W1     = (const float*)d_in[3];
    const float* b1     = (const float*)d_in[4];
    const float* W2     = (const float*)d_in[5];
    const float* b2     = (const float*)d_in[6];
    const float* W3     = (const float*)d_in[7];
    const float* b3     = (const float*)d_in[8];

    float* outx  = (float*)d_out;
    float* outld = outx + (size_t)Bn * Cc * Hh * Ww;

    char* ws = (char*)d_ws;
    _Float16* zg  = (_Float16*)(ws);                     // 8,388,608 B
    _Float16* h2g = (_Float16*)(ws + 8388608);           // 33,554,432 B
    _Float16* W1p = (_Float16*)(ws + 8388608 + 33554432);              // 20,480 B
    _Float16* W2p = (_Float16*)(ws + 8388608 + 33554432 + 20480);      //  8,192 B
    _Float16* W3p = (_Float16*)(ws + 8388608 + 33554432 + 20480 + 8192); // 331,776 B

    int packN = HID * K1P + HID * HID + MPAD * K3;       // 180224
    pack_weights_kernel<<<(packN + 255) / 256, 256, 0, stream>>>(
        W1, W2, W3, logdet, W1p, W2p, W3p, outld);
    pack_z_kernel<<<(Bn * Hh * Ww * CIN1) / 256, 256, 0, stream>>>(x, cond, zg);
    conv12_kernel<<<Bn * Hh * 2, C12T, 0, stream>>>(zg, W1p, b1, W2p, b2, h2g);
    conv3_rqs_kernel<<<Bn * Hh * 2, C3T, 0, stream>>>(h2g, W3p, b3, x, outx, outld);
}
